// PersistentGRUCell_11587821765036
// MI455X (gfx1250) — matrix operand order repaired and checked by reading, not compiled, not run on hardware
//
#include <hip/hip_runtime.h>
#include <hip/hip_bf16.h>
#include <math.h>

// ---------------------------------------------------------------------------
// Types for CDNA5 WMMA (wave32): v_wmma_f32_16x16x32_bf16
// ---------------------------------------------------------------------------
typedef __bf16 bf16_t;
typedef __bf16 v16bf __attribute__((ext_vector_type(16)));
typedef __bf16 v8bf  __attribute__((ext_vector_type(8)));
typedef float  v8f   __attribute__((ext_vector_type(8)));

union BFrag {
  v16bf v;
  v8bf  h[2];
};

__device__ __forceinline__ float sigmoidf_(float x) {
  return 1.0f / (1.0f + __expf(-x));
}

// ---------------------------------------------------------------------------
// f32 -> bf16 conversion (grid-stride)
// ---------------------------------------------------------------------------
__global__ void k_f32_to_bf16(const float* __restrict__ s, bf16_t* __restrict__ d, long n) {
  long i = (long)blockIdx.x * blockDim.x + threadIdx.x;
  long st = (long)gridDim.x * blockDim.x;
  for (; i < n; i += st) d[i] = (bf16_t)s[i];
}

// ---------------------------------------------------------------------------
// activity[h] += sum_{rows in chunk} |hidden[b,h]|   (act must be zeroed first)
// ---------------------------------------------------------------------------
__global__ void k_colabs_partial(const float* __restrict__ h, float* __restrict__ act,
                                 int H, int rowsPerChunk) {
  int col = blockIdx.x * blockDim.x + threadIdx.x;
  int r0  = blockIdx.y * rowsPerChunk;
  float s = 0.f;
  for (int b = 0; b < rowsPerChunk; ++b) s += fabsf(h[(size_t)(r0 + b) * H + col]);
  atomicAdd(&act[col], s);
}

__global__ void k_cs(const float* __restrict__ act, const float* __restrict__ strength,
                     const float* __restrict__ decay, const float* __restrict__ history,
                     float* __restrict__ cs, int H, float invB) {
  int hI = blockIdx.x * blockDim.x + threadIdx.x;
  if (hI < H) {
    float ch = history[hI] * decay[0] + act[hI] * invB;
    cs[hI] = strength[hI] * sigmoidf_(ch);
  }
}

// ---------------------------------------------------------------------------
// WMMA GEMM:  C[row,col] = epilogue( sum_k A[row,k] * W[col,k] + bias[col] )
//   A: bf16 row-major, lda = 1024 (compile-time), split at Ksplit between
//      A0 and A1 to model concat([x, ha]) without materializing it.
//   W: bf16 row-major [N, LDW] (LDW compile-time).
// Each wave computes a 64x64 tile (4 M-frags x 4 N-frags), K step = 32.
// All fragment loads are base + immediate-offset b128s; only two 64-bit
// pointer bumps per K iteration.
// Fragment layouts follow CDNA5 ISA 7.12.2 (wave32).
// ---------------------------------------------------------------------------
struct EpArgs {
  const float*  f0;   // hidden (ATT/NEW) / pm (QUAD)
  const float*  f1;   // upd (NEW) / hp (QUAD)
  const float*  f2;   // cs (NEW)
  const bf16_t* bin;  // ha_b (RESET)
  bf16_t*       bout; // ha_b (ATT) / rha_b (RESET) / hn_b (NEW)
  float*        o0;   // upd (UPD) / hn_f (NEW) / r_f (QUAD seg0)
  float*        o1;   // wcoef (QUAD seg1)
  float*        o2;   // base  (QUAD seg2)
  float*        o3;   // addr logits (QUAD seg3)
};

enum { EP_ATT = 0, EP_RESET = 1, EP_UPD = 2, EP_NEW = 3, EP_QUAD = 4 };

template <int EP, int LDW>
__global__ __launch_bounds__(128)
void k_gemm(const bf16_t* __restrict__ A0, const bf16_t* __restrict__ A1, int Ksplit,
            const bf16_t* __restrict__ W, const float* __restrict__ bias,
            int N, int K, EpArgs ep) {
  constexpr int LDA = 1024;
  const int lane = threadIdx.x & 31;
  const int half = lane >> 4;   // 0/1: which 16-lane group
  const int lr   = lane & 15;
  const int wid  = blockIdx.x * (blockDim.x >> 5) + (threadIdx.x >> 5);
  const int nWN  = N >> 6;
  const int waveM = wid / nWN;
  const int waveN = wid % nWN;
  const int rowBase = waveM << 6;   // 64 rows per wave
  const int colBase = waveN << 6;   // 64 cols per wave

  v8f acc[4][4] = {};

  // B fragments (32x16 KxN): lane holds column N=lr; elems 0..7 at k+8*half, elems 8..15 at k+16+8*half.
  const bf16_t* wbase = W + (size_t)(colBase + lr) * LDW + (half << 3);

  auto kloop = [&](const bf16_t* __restrict__ Aseg, int kcount) {
    // A fragments (16x32 MxK): lane holds row M=lr; elems 0..7 at k+8*half,
    // elems 8..15 at k+16+8*half  (ISA 7.12.2, 16-bit A 16x32)
    const bf16_t* abase = Aseg + (size_t)(rowBase + lr) * LDA + (half << 3);
    for (int k = 0; k < kcount; k += 32) {
      BFrag fa[4], fb[4];
#pragma unroll
      for (int mt = 0; mt < 4; ++mt) {
        fa[mt].h[0] = *(const v8bf*)(abase + mt * 16 * LDA);
        fa[mt].h[1] = *(const v8bf*)(abase + mt * 16 * LDA + 16);
      }
#pragma unroll
      for (int nt = 0; nt < 4; ++nt) {
        fb[nt].h[0] = *(const v8bf*)(wbase + nt * 16 * LDW);
        fb[nt].h[1] = *(const v8bf*)(wbase + nt * 16 * LDW + 16);
      }
#pragma unroll
      for (int mt = 0; mt < 4; ++mt)
#pragma unroll
        for (int nt = 0; nt < 4; ++nt)
          acc[mt][nt] = __builtin_amdgcn_wmma_f32_16x16x32_bf16(
              false, fa[mt].v, false, fb[nt].v, (short)0, acc[mt][nt], false, false);
      abase += 32;
      wbase += 32;   // W has full K columns; advances across both segments
    }
  };
  kloop(A0, Ksplit);
  if (Ksplit < K) kloop(A1, K - Ksplit);

  // C layout: VGPR r holds M = r + 8*half, N = lr  (ISA 7.12.2, 32-bit C 16x16)
#pragma unroll
  for (int mt = 0; mt < 4; ++mt) {
    const int rb = rowBase + (mt << 4) + (half << 3);
#pragma unroll
    for (int nt = 0; nt < 4; ++nt) {
      const int col = colBase + (nt << 4) + lr;
      const float bcol = bias[col];
#pragma unroll
      for (int r = 0; r < 8; ++r) {
        const int row = rb + r;
        float v = acc[mt][nt][r] + bcol;
        if constexpr (EP == EP_ATT) {
          const size_t idx = (size_t)row * N + col;
          ep.bout[idx] = (bf16_t)(ep.f0[idx] * sigmoidf_(v));      // ha = h * sigm(att)
        } else if constexpr (EP == EP_RESET) {
          const size_t idx = (size_t)row * N + col;
          ep.bout[idx] = (bf16_t)(v * (float)ep.bin[idx]);         // reset * ha
        } else if constexpr (EP == EP_UPD) {
          ep.o0[(size_t)row * N + col] = v;                        // update (raw)
        } else if constexpr (EP == EP_NEW) {
          const size_t idx = (size_t)row * N + col;
          const float u  = ep.f1[idx];
          const float hn = (1.0f - u) * ep.f0[idx] + u * v * ep.f2[col];
          ep.o0[idx]  = hn;
          ep.bout[idx] = (bf16_t)hn;
        } else {  // EP_QUAD: col segment selects read/write/forget/addr
          const int seg = col >> 11;
          const int p   = col & 2047;
          const size_t idx = (size_t)row * 2048 + p;
          if (seg == 3) {
            ep.o3[idx] = v;                                        // addr logits
          } else {
            const float s = sigmoidf_(v);
            if (seg == 0)      ep.o0[idx] = s;                     // read
            else if (seg == 1) ep.o1[idx] = s * ep.f1[row];        // write * hp[b]
            else               ep.o2[idx] = s * ep.f0[idx];        // forget * pm
          }
        }
      }
    }
  }
}

// ---------------------------------------------------------------------------
// hp[b] = mean over H of x[b,:]
// ---------------------------------------------------------------------------
__global__ void k_rowmean(const float* __restrict__ x, float* __restrict__ out, int H) {
  __shared__ float sm[256];
  const size_t b = blockIdx.x;
  float s = 0.f;
  for (int hI = threadIdx.x; hI < H; hI += 256) s += x[b * H + hI];
  sm[threadIdx.x] = s;
  __syncthreads();
  for (int o = 128; o > 0; o >>= 1) {
    if (threadIdx.x < (unsigned)o) sm[threadIdx.x] += sm[threadIdx.x + o];
    __syncthreads();
  }
  if (threadIdx.x == 0) out[b] = sm[0] / (float)H;
}

// ---------------------------------------------------------------------------
// Per-row: addr = softmax(L); mem = base + wcoef*addr;  out_m = mem;
// out_h = hn + mean_P(read * mem)
// ---------------------------------------------------------------------------
__global__ void k_finalize(const float* __restrict__ L, const float* __restrict__ base,
                           const float* __restrict__ wc, const float* __restrict__ rf,
                           const float* __restrict__ hn,
                           float* __restrict__ outH, float* __restrict__ outM,
                           int H, int P) {
  __shared__ float sm[256];
  __shared__ float sval;
  const size_t b = blockIdx.x;
  const int t = threadIdx.x;

  // row max
  float mx = -3.4e38f;
  for (int p = t; p < P; p += 256) mx = fmaxf(mx, L[b * P + p]);
  sm[t] = mx;
  __syncthreads();
  for (int o = 128; o > 0; o >>= 1) { if (t < o) sm[t] = fmaxf(sm[t], sm[t + o]); __syncthreads(); }
  if (t == 0) sval = sm[0];
  __syncthreads();
  mx = sval;
  __syncthreads();

  // sum exp
  float se = 0.f;
  for (int p = t; p < P; p += 256) se += __expf(L[b * P + p] - mx);
  sm[t] = se;
  __syncthreads();
  for (int o = 128; o > 0; o >>= 1) { if (t < o) sm[t] += sm[t + o]; __syncthreads(); }
  if (t == 0) sval = sm[0];
  __syncthreads();
  const float inv = 1.0f / sval;
  __syncthreads();

  // memory_new + read accumulation
  float a = 0.f;
  for (int p = t; p < P; p += 256) {
    const size_t i = b * P + p;
    const float addr = __expf(L[i] - mx) * inv;
    const float m = base[i] + wc[i] * addr;
    outM[i] = m;
    a += rf[i] * m;
  }
  sm[t] = a;
  __syncthreads();
  for (int o = 128; o > 0; o >>= 1) { if (t < o) sm[t] += sm[t + o]; __syncthreads(); }
  if (t == 0) sval = sm[0] / (float)P;
  __syncthreads();
  const float add = sval;
  for (int hI = t; hI < H; hI += 256) outH[b * H + hI] = hn[b * H + hI] + add;
}

// ---------------------------------------------------------------------------
// Launch
// ---------------------------------------------------------------------------
extern "C" void kernel_launch(void* const* d_in, const int* in_sizes, int n_in,
                              void* d_out, int out_size, void* d_ws, size_t ws_size,
                              hipStream_t stream) {
  (void)in_sizes; (void)n_in; (void)out_size; (void)ws_size;
  constexpr int B = 8192, I = 1024, H = 1024, P = 2048;
  constexpr int K2 = I + H;        // 2048
  constexpr int N4 = 4 * P;        // 8192 (read|write|forget|addr packed)

  const float* x      = (const float*)d_in[0];
  const float* hidden = (const float*)d_in[1];
  const float* pm     = (const float*)d_in[2];
  const float* Wa     = (const float*)d_in[3];
  const float* ba     = (const float*)d_in[4];
  const float* Wr     = (const float*)d_in[5];
  const float* br     = (const float*)d_in[6];
  const float* Wu     = (const float*)d_in[7];
  const float* bu     = (const float*)d_in[8];
  const float* Wn     = (const float*)d_in[9];
  const float* bn     = (const float*)d_in[10];
  const float* Wread  = (const float*)d_in[11];
  const float* bread  = (const float*)d_in[12];
  const float* Wwrite = (const float*)d_in[13];
  const float* bwrite = (const float*)d_in[14];
  const float* Wforget= (const float*)d_in[15];
  const float* bforget= (const float*)d_in[16];
  const float* Waddr  = (const float*)d_in[17];
  const float* baddr  = (const float*)d_in[18];
  const float* cstr   = (const float*)d_in[19];
  const float* cdec   = (const float*)d_in[20];
  const float* chist  = (const float*)d_in[21];

  float* outH = (float*)d_out;                       // hidden_final [B,H]
  float* outM = outH + (size_t)B * H;                // memory_new   [B,P]

  // ---- workspace bump allocator (256B aligned) ----
  size_t off = 0;
  auto alloc = [&](size_t nbytes) -> void* {
    void* p = (char*)d_ws + off;
    off += (nbytes + 255) & ~(size_t)255;
    return p;
  };
  float*  act    = (float*)alloc(H * 4);
  float*  cs     = (float*)alloc(H * 4);
  float*  hp     = (float*)alloc(B * 4);
  float*  b4     = (float*)alloc(N4 * 4);
  bf16_t* hb     = (bf16_t*)alloc((size_t)B * H * 2);
  bf16_t* xb     = (bf16_t*)alloc((size_t)B * I * 2);
  bf16_t* Wa_b   = (bf16_t*)alloc((size_t)H * H * 2);
  bf16_t* Wr_b   = (bf16_t*)alloc((size_t)H * K2 * 2);
  bf16_t* Wu_b   = (bf16_t*)alloc((size_t)H * K2 * 2);
  bf16_t* Wn_b   = (bf16_t*)alloc((size_t)H * K2 * 2);
  bf16_t* W4_b   = (bf16_t*)alloc((size_t)N4 * H * 2);
  bf16_t* ha_b   = (bf16_t*)alloc((size_t)B * H * 2);
  bf16_t* rha_b  = (bf16_t*)alloc((size_t)B * H * 2);
  float*  upd    = (float*)alloc((size_t)B * H * 4);
  float*  hn_f   = (float*)alloc((size_t)B * H * 4);
  bf16_t* hn_b   = (bf16_t*)alloc((size_t)B * H * 2);
  float*  r_f    = (float*)alloc((size_t)B * P * 4);
  float*  wcoef  = (float*)alloc((size_t)B * P * 4);
  float*  baseb  = (float*)alloc((size_t)B * P * 4);
  float*  Lbuf   = (float*)alloc((size_t)B * P * 4);

  // ---- conversions to bf16 ----
  const int CB = 2048, CT = 256;
  k_f32_to_bf16<<<CB, CT, 0, stream>>>(hidden, hb, (long)B * H);
  k_f32_to_bf16<<<CB, CT, 0, stream>>>(x, xb, (long)B * I);
  k_f32_to_bf16<<<CB, CT, 0, stream>>>(Wa, Wa_b, (long)H * H);
  k_f32_to_bf16<<<CB, CT, 0, stream>>>(Wr, Wr_b, (long)H * K2);
  k_f32_to_bf16<<<CB, CT, 0, stream>>>(Wu, Wu_b, (long)H * K2);
  k_f32_to_bf16<<<CB, CT, 0, stream>>>(Wn, Wn_b, (long)H * K2);
  k_f32_to_bf16<<<CB, CT, 0, stream>>>(Wread,  W4_b + (size_t)0 * P * H, (long)P * H);
  k_f32_to_bf16<<<CB, CT, 0, stream>>>(Wwrite, W4_b + (size_t)1 * P * H, (long)P * H);
  k_f32_to_bf16<<<CB, CT, 0, stream>>>(Wforget,W4_b + (size_t)2 * P * H, (long)P * H);
  k_f32_to_bf16<<<CB, CT, 0, stream>>>(Waddr,  W4_b + (size_t)3 * P * H, (long)P * H);

  // packed bias for the fused quad GEMM
  hipMemcpyAsync(b4 + 0 * P, bread,   P * 4, hipMemcpyDeviceToDevice, stream);
  hipMemcpyAsync(b4 + 1 * P, bwrite,  P * 4, hipMemcpyDeviceToDevice, stream);
  hipMemcpyAsync(b4 + 2 * P, bforget, P * 4, hipMemcpyDeviceToDevice, stream);
  hipMemcpyAsync(b4 + 3 * P, baddr,   P * 4, hipMemcpyDeviceToDevice, stream);

  // ---- connection strength ----
  hipMemsetAsync(act, 0, H * 4, stream);
  {
    dim3 g(H / 256, 32);
    k_colabs_partial<<<g, 256, 0, stream>>>(hidden, act, H, B / 32);
  }
  k_cs<<<H / 256, 256, 0, stream>>>(act, cstr, cdec, chist, cs, H, 1.0f / (float)B);

  // ---- GEMM chain (64x64 tile per wave, 4 waves per block) ----
  auto gemm_blocks = [](int n) { return (B / 64) * (n / 64) / 4; };

  { EpArgs e{}; e.f0 = hidden; e.bout = ha_b;
    k_gemm<EP_ATT, 1024><<<gemm_blocks(H), 128, 0, stream>>>(hb, hb, H, Wa_b, ba, H, H, e); }

  { EpArgs e{}; e.bin = ha_b; e.bout = rha_b;
    k_gemm<EP_RESET, 2048><<<gemm_blocks(H), 128, 0, stream>>>(xb, ha_b, I, Wr_b, br, H, K2, e); }

  { EpArgs e{}; e.o0 = upd;
    k_gemm<EP_UPD, 2048><<<gemm_blocks(H), 128, 0, stream>>>(xb, ha_b, I, Wu_b, bu, H, K2, e); }

  { EpArgs e{}; e.f0 = hidden; e.f1 = upd; e.f2 = cs; e.o0 = hn_f; e.bout = hn_b;
    k_gemm<EP_NEW, 2048><<<gemm_blocks(H), 128, 0, stream>>>(xb, rha_b, I, Wn_b, bn, H, K2, e); }

  k_rowmean<<<B, 256, 0, stream>>>(hn_f, hp, H);

  { EpArgs e{}; e.f0 = pm; e.f1 = hp; e.o0 = r_f; e.o1 = wcoef; e.o2 = baseb; e.o3 = Lbuf;
    k_gemm<EP_QUAD, 1024><<<gemm_blocks(N4), 128, 0, stream>>>(hn_b, hn_b, H, W4_b, b4, N4, H, e); }

  // ---- softmax + memory update + read mean ----
  k_finalize<<<B, 256, 0, stream>>>(Lbuf, baseb, wcoef, r_f, hn_f, outH, outM, H, P);
}
